// NER_CRF_89223650607182
// MI455X (gfx1250) — matrix operand order repaired and checked by reading, not compiled, not run on hardware
//
#include <hip/hip_runtime.h>
#include <hip/hip_bf16.h>

typedef __attribute__((ext_vector_type(16))) _Float16 v16h;
typedef __attribute__((ext_vector_type(8)))  float    v8f;

// ---------------------------------------------------------------------------
// Async global->LDS 16-byte copy (CDNA5 GLOBAL_LOAD_ASYNC_TO_LDS_B128,
// tracked by ASYNCcnt). VDST VGPR holds the wave-relative LDS byte address:
// for a generic pointer derived from a __shared__ object the low 32 bits are
// the AS(3) offset (LDS aperture keeps the offset in addr[31:0]).
// ---------------------------------------------------------------------------
__device__ __forceinline__ void async_copy16(void* lds_dst, const void* gsrc) {
  unsigned lds_off = (unsigned)(unsigned long long)lds_dst;
  asm volatile("global_load_async_to_lds_b128 %0, %1, off"
               :: "v"(lds_off), "v"(gsrc) : "memory");
}
__device__ __forceinline__ void async_wait0() {
  asm volatile("s_wait_asynccnt 0" ::: "memory");
}

// ---------------------------------------------------------------------------
// WMMA fragment loaders for V_WMMA_F32_16X16X32_F16 (wave32).
// A (16x32, MxK), this code's contiguous k order: lane L holds row M=(L&15);
//   halves i (0..15) -> K = k0 + (L>=16 ? 16 : 0) + i  (same order as B)
// B (32x16, KxN): lane L holds col N=(L&15); halves i -> K = k0 + (L>=16?16:0)+i.
// B comes from a transposed [N][K] f16 image (global or LDS) so it's contiguous.
// ---------------------------------------------------------------------------
__device__ __forceinline__ v16h frag_a(const _Float16* A, int lda, int row0,
                                       int k0, int lane) {
  const _Float16* p = A + (row0 + (lane & 15)) * lda + k0 + ((lane & 16) ? 16 : 0);
  v16h a;
#pragma unroll
  for (int i = 0; i < 8; ++i) { a[i] = p[i]; a[i + 8] = p[i + 8]; }
  return a;
}

__device__ __forceinline__ v16h frag_b(const _Float16* WT, int ldk, int col0,
                                       int k0, int lane) {
  const _Float16* p = WT + (col0 + (lane & 15)) * ldk + k0 + ((lane & 16) ? 16 : 0);
  v16h b;
#pragma unroll
  for (int i = 0; i < 16; ++i) b[i] = p[i];
  return b;
}

__device__ __forceinline__ float sigf(float x) { return 1.f / (1.f + __expf(-x)); }

// ---------------------------------------------------------------------------
// K0: float [K][N] -> f16 transposed [N][K]
// ---------------------------------------------------------------------------
__global__ void cvt_transpose_kernel(const float* __restrict__ in,
                                     _Float16* __restrict__ out, int K, int N) {
  int idx = blockIdx.x * blockDim.x + threadIdx.x;
  if (idx >= K * N) return;
  int n = idx / K, k = idx % K;
  out[idx] = (_Float16)in[k * N + n];
}

// ---------------------------------------------------------------------------
// K1: embedding gather -> f16 X[16384][256]
// ---------------------------------------------------------------------------
__global__ void gather_kernel(const int* __restrict__ ids,
                              const float* __restrict__ emb,
                              _Float16* __restrict__ X) {
  int idx = blockIdx.x * blockDim.x + threadIdx.x;   // 16384*256 exact
  int i = idx >> 8, e = idx & 255;
  X[idx] = (_Float16)emb[ids[i] * 256 + e];
}

// ---------------------------------------------------------------------------
// K2: XG = X @ wx + b  (M=16384, K=256, N=1024), both directions via grid.z.
// 8 waves/block share one 64-col weight panel staged once in LDS (32 KB) via
// async global->LDS copies (no VGPR round-trip); B fragments are ds_load_b128.
// Output stored TIME-MAJOR: XGt[t][col][b]  (t=token&511, b=token>>9) so the
// serial LSTM can read its gate inputs as contiguous b128 loads.
// ---------------------------------------------------------------------------
__global__ __launch_bounds__(256) void gemm_xg_kernel(
    const _Float16* __restrict__ X,
    const _Float16* __restrict__ wxTf, const float* __restrict__ bf,
    const _Float16* __restrict__ wxTb, const float* __restrict__ bb,
    float* __restrict__ XGtf, float* __restrict__ XGtb) {
  const int lane = threadIdx.x & 31;
  const int w    = threadIdx.x >> 5;                 // 0..7
  const int row0 = (blockIdx.x * 8 + w) * 16;
  const int col0 = blockIdx.y * 64;
  const _Float16* WT = blockIdx.z ? wxTb : wxTf;
  const float* bias  = blockIdx.z ? bb : bf;
  float* XGt         = blockIdx.z ? XGtb : XGtf;

  __shared__ alignas(16) _Float16 wpanel[64 * 256];  // 32 KB, [n][k] n-local
  {
    const float4* src = (const float4*)(WT + (size_t)col0 * 256);  // contiguous
    float4* dst = (float4*)wpanel;
    for (int c = threadIdx.x; c < 2048; c += 256) async_copy16(dst + c, src + c);
    async_wait0();
  }
  __syncthreads();

  v8f acc[4];
#pragma unroll
  for (int g = 0; g < 4; ++g)
#pragma unroll
    for (int v = 0; v < 8; ++v) acc[g][v] = 0.f;

  for (int k = 0; k < 256; k += 32) {
    v16h a  = frag_a(X, 256, row0, k, lane);
    v16h b0 = frag_b(wpanel, 256,  0, k, lane);
    v16h b1 = frag_b(wpanel, 256, 16, k, lane);
    v16h b2 = frag_b(wpanel, 256, 32, k, lane);
    v16h b3 = frag_b(wpanel, 256, 48, k, lane);
    acc[0] = __builtin_amdgcn_wmma_f32_16x16x32_f16(false, a, false, b0, (short)0, acc[0], false, false);
    acc[1] = __builtin_amdgcn_wmma_f32_16x16x32_f16(false, a, false, b1, (short)0, acc[1], false, false);
    acc[2] = __builtin_amdgcn_wmma_f32_16x16x32_f16(false, a, false, b2, (short)0, acc[2], false, false);
    acc[3] = __builtin_amdgcn_wmma_f32_16x16x32_f16(false, a, false, b3, (short)0, acc[3], false, false);
  }

  const int rofs = (lane & 16) ? 8 : 0;
#pragma unroll
  for (int g = 0; g < 4; ++g)
#pragma unroll
    for (int v = 0; v < 8; ++v) {
      int row = row0 + rofs + v;                // token index
      int col = col0 + g * 16 + (lane & 15);
      int t = row & 511, bidx = row >> 9;
      XGt[((size_t)(t * 1024 + col)) * 32 + bidx] = acc[g][v] + bias[col];
    }
}

// ---------------------------------------------------------------------------
// K3: persistent-workgroup LSTM. grid.x = direction (0 fwd, 1 bwd).
// 32 waves: wave w owns rows mt..mt+15 (mt=(w&1)*16) and H-columns
// jt..jt+15 (jt=(w>>1)*16) across all 4 gate blocks, so cell state c stays
// in v8f registers for the whole scan; h lives in LDS as f16.
// Gate inputs come from the time-major XGt as contiguous v8f loads.
// ---------------------------------------------------------------------------
__global__ __launch_bounds__(1024) void lstm_kernel(
    const float* __restrict__ XGtf, const float* __restrict__ XGtb,
    const _Float16* __restrict__ whTf, const _Float16* __restrict__ whTb,
    _Float16* __restrict__ feats) {
  const int dir = blockIdx.x;
  const float* XG    = dir ? XGtb : XGtf;
  const _Float16* WT = dir ? whTb : whTf;
  const int lane = threadIdx.x & 31;
  const int w    = threadIdx.x >> 5;   // 0..31
  const int mt   = (w & 1) * 16;
  const int jt   = (w >> 1) * 16;
  const int rofs = (lane & 16) ? 8 : 0;
  const int nlan = lane & 15;

  __shared__ _Float16 hbuf[32 * 256];
  for (int i = threadIdx.x; i < 32 * 256; i += 1024) hbuf[i] = (_Float16)0.f;

  v8f c;
#pragma unroll
  for (int v = 0; v < 8; ++v) c[v] = 0.f;
  __syncthreads();

  for (int step = 0; step < 512; ++step) {
    const int t = dir ? (511 - step) : step;

    // init accumulators with precomputed xg (bias included); contiguous loads
    v8f acc[4];
#pragma unroll
    for (int g = 0; g < 4; ++g) {
      const float* p = XG + ((size_t)(t * 1024 + g * 256 + jt + nlan)) * 32 + mt + rofs;
#pragma unroll
      for (int v = 0; v < 8; ++v) acc[g][v] = p[v];
    }

    if (step + 1 < 512) {   // pull next step's gate inputs toward the WGP
      int tn = dir ? (510 - step) : (step + 1);
      __builtin_prefetch(XG + ((size_t)(tn * 1024 + jt + nlan)) * 32 + mt + rofs, 0, 0);
    }

    // g += h @ wh
    for (int k = 0; k < 256; k += 32) {
      v16h a  = frag_a(hbuf, 256, mt, k, lane);
      v16h b0 = frag_b(WT, 256, 0 * 256 + jt, k, lane);
      v16h b1 = frag_b(WT, 256, 1 * 256 + jt, k, lane);
      v16h b2 = frag_b(WT, 256, 2 * 256 + jt, k, lane);
      v16h b3 = frag_b(WT, 256, 3 * 256 + jt, k, lane);
      acc[0] = __builtin_amdgcn_wmma_f32_16x16x32_f16(false, a, false, b0, (short)0, acc[0], false, false);
      acc[1] = __builtin_amdgcn_wmma_f32_16x16x32_f16(false, a, false, b1, (short)0, acc[1], false, false);
      acc[2] = __builtin_amdgcn_wmma_f32_16x16x32_f16(false, a, false, b2, (short)0, acc[2], false, false);
      acc[3] = __builtin_amdgcn_wmma_f32_16x16x32_f16(false, a, false, b3, (short)0, acc[3], false, false);
    }
    __syncthreads();  // everyone done reading old h

#pragma unroll
    for (int v = 0; v < 8; ++v) {
      float ig = sigf(acc[0][v]);
      float fg = sigf(acc[1][v]);
      float gg = tanhf(acc[2][v]);
      float og = sigf(acc[3][v]);
      float cn = fg * c[v] + ig * gg;
      c[v] = cn;
      float hh = og * tanhf(cn);
      int b   = mt + rofs + v;
      int col = jt + nlan;
      hbuf[b * 256 + col] = (_Float16)hh;
      feats[(b * 512 + t) * 512 + dir * 256 + col] = (_Float16)hh;
    }
    __syncthreads();  // new h visible before next step's reads
  }
}

// ---------------------------------------------------------------------------
// K4: hidden = relu(feats @ w1 + b1)   (M=16384, K=512, N=512) -> f16.
// Weight panel staged in LDS in two 32 KB half-K passes via async copies.
// ---------------------------------------------------------------------------
__global__ __launch_bounds__(256) void mlp_kernel(const _Float16* __restrict__ F,
                                                  const _Float16* __restrict__ w1T,
                                                  const float* __restrict__ b1,
                                                  _Float16* __restrict__ Hd) {
  const int lane = threadIdx.x & 31;
  const int w    = threadIdx.x >> 5;
  const int row0 = (blockIdx.x * 8 + w) * 16;
  const int col0 = blockIdx.y * 64;

  __shared__ alignas(16) _Float16 wpanel[64 * 256];  // 32 KB half-K panel

  v8f acc[4];
#pragma unroll
  for (int g = 0; g < 4; ++g)
#pragma unroll
    for (int v = 0; v < 8; ++v) acc[g][v] = 0.f;

  for (int kh = 0; kh < 512; kh += 256) {
    __syncthreads();  // previous panel fully consumed
    {
      float4* dst = (float4*)wpanel;
      for (int ch = threadIdx.x; ch < 2048; ch += 256) {
        int n = ch >> 5, j = ch & 31;   // 32 float4 per 256-half row
        async_copy16(dst + ch, (const float4*)(w1T + (size_t)(col0 + n) * 512 + kh) + j);
      }
      async_wait0();
    }
    __syncthreads();

    for (int k = 0; k < 256; k += 32) {
      v16h a  = frag_a(F, 512, row0, kh + k, lane);
      v16h b0 = frag_b(wpanel, 256,  0, k, lane);
      v16h b1 = frag_b(wpanel, 256, 16, k, lane);
      v16h b2 = frag_b(wpanel, 256, 32, k, lane);
      v16h b3 = frag_b(wpanel, 256, 48, k, lane);
      acc[0] = __builtin_amdgcn_wmma_f32_16x16x32_f16(false, a, false, b0, (short)0, acc[0], false, false);
      acc[1] = __builtin_amdgcn_wmma_f32_16x16x32_f16(false, a, false, b1, (short)0, acc[1], false, false);
      acc[2] = __builtin_amdgcn_wmma_f32_16x16x32_f16(false, a, false, b2, (short)0, acc[2], false, false);
      acc[3] = __builtin_amdgcn_wmma_f32_16x16x32_f16(false, a, false, b3, (short)0, acc[3], false, false);
    }
  }

  const int rofs = (lane & 16) ? 8 : 0;
#pragma unroll
  for (int g = 0; g < 4; ++g)
#pragma unroll
    for (int v = 0; v < 8; ++v) {
      int row = row0 + rofs + v;
      int col = col0 + g * 16 + (lane & 15);
      float val = acc[g][v] + b1[col];
      Hd[row * 512 + col] = (_Float16)(val > 0.f ? val : 0.f);
    }
}

// ---------------------------------------------------------------------------
// K5: logits = hidden @ w2 + b2   (N = 11, scalar dots)
// ---------------------------------------------------------------------------
__global__ void logits_kernel(const _Float16* __restrict__ Hd,
                              const float* __restrict__ w2,
                              const float* __restrict__ b2,
                              float* __restrict__ out) {
  int idx = blockIdx.x * blockDim.x + threadIdx.x;
  if (idx >= 16384 * 11) return;
  int i = idx / 11, tag = idx % 11;
  float s = b2[tag];
  const _Float16* hp = Hd + i * 512;
  for (int k = 0; k < 512; ++k) s += (float)hp[k] * w2[k * 11 + tag];
  out[idx] = s;
}

// ---------------------------------------------------------------------------
// K6: CRF negative log-likelihood. One block, one thread per (batch, tag).
// ---------------------------------------------------------------------------
__global__ __launch_bounds__(384) void crf_kernel(const float* __restrict__ logits,
                                                  const int* __restrict__ mask,
                                                  const int* __restrict__ labels,
                                                  const float* __restrict__ trans,
                                                  float* __restrict__ loss_out) {
  __shared__ float tr[11][11];
  __shared__ float alpha[32][12];
  __shared__ float logz[32];
  __shared__ float gold[32];
  const int tid = threadIdx.x;
  if (tid < 121) tr[tid / 11][tid % 11] = trans[tid];
  __syncthreads();

  const int b = tid / 11, j = tid % 11;
  const bool act = tid < 352;
  if (act) alpha[b][j] = tr[9][j] + logits[(b * 512) * 11 + j];  // START=9
  __syncthreads();

  for (int t = 1; t < 512; ++t) {
    float nv = 0.f;
    if (act) {
      float m = -1e30f;
#pragma unroll
      for (int i = 0; i < 11; ++i) m = fmaxf(m, alpha[b][i] + tr[i][j]);
      float s = 0.f;
#pragma unroll
      for (int i = 0; i < 11; ++i) s += __expf(alpha[b][i] + tr[i][j] - m);
      float lse = m + __logf(s) + logits[(b * 512 + t) * 11 + j];
      nv = (mask[b * 512 + t] > 0) ? lse : alpha[b][j];
    }
    __syncthreads();
    if (act) alpha[b][j] = nv;
    __syncthreads();
  }

  if (act && j == 0) {                 // STOP=10
    float m = -1e30f;
    for (int i = 0; i < 11; ++i) m = fmaxf(m, alpha[b][i] + tr[i][10]);
    float s = 0.f;
    for (int i = 0; i < 11; ++i) s += __expf(alpha[b][i] + tr[i][10] - m);
    logz[b] = m + __logf(s);
  }
  if (tid < 32) {
    const int bb = tid;
    float e = 0.f, tsc = 0.f;
    int len = 0, prev = 0;
    for (int t = 0; t < 512; ++t) {
      int mk = mask[bb * 512 + t];
      int lab = labels[bb * 512 + t];
      float mf = (float)mk;
      e += logits[(bb * 512 + t) * 11 + lab] * mf;
      if (t > 0) tsc += tr[prev][lab] * mf;
      len += (mk != 0);
      prev = lab;
    }
    int last = labels[bb * 512 + (len > 0 ? len - 1 : 0)];
    gold[bb] = e + tr[9][labels[bb * 512]] + tsc + tr[last][10];
  }
  __syncthreads();
  if (tid == 0) {
    float s = 0.f;
    for (int i = 0; i < 32; ++i) s += logz[i] - gold[i];
    loss_out[0] = s / 32.f;
  }
}

// ---------------------------------------------------------------------------
extern "C" void kernel_launch(void* const* d_in, const int* in_sizes, int n_in,
                              void* d_out, int out_size, void* d_ws, size_t ws_size,
                              hipStream_t stream) {
  const int*   input_ids = (const int*)  d_in[0];
  const int*   attn_mask = (const int*)  d_in[1];
  const int*   labels    = (const int*)  d_in[2];
  const float* emb       = (const float*)d_in[3];
  const float* wx_f      = (const float*)d_in[4];
  const float* wh_f      = (const float*)d_in[5];
  const float* b_f       = (const float*)d_in[6];
  const float* wx_b      = (const float*)d_in[7];
  const float* wh_b      = (const float*)d_in[8];
  const float* b_b       = (const float*)d_in[9];
  const float* w1        = (const float*)d_in[10];
  const float* b1        = (const float*)d_in[11];
  const float* w2        = (const float*)d_in[12];
  const float* b2        = (const float*)d_in[13];
  const float* trans     = (const float*)d_in[14];

  char* ws = (char*)d_ws;
  size_t off = 0;
  auto alloc = [&](size_t bytes) -> void* {
    void* p = ws + off;
    off += (bytes + 255) & ~(size_t)255;
    return p;
  };
  _Float16* wxfT  = (_Float16*)alloc((size_t)1024 * 256 * 2);
  _Float16* wxbT  = (_Float16*)alloc((size_t)1024 * 256 * 2);
  _Float16* whfT  = (_Float16*)alloc((size_t)1024 * 256 * 2);
  _Float16* whbT  = (_Float16*)alloc((size_t)1024 * 256 * 2);
  _Float16* w1T   = (_Float16*)alloc((size_t)512 * 512 * 2);
  _Float16* X     = (_Float16*)alloc((size_t)16384 * 256 * 2);
  float*    XGtf  = (float*)   alloc((size_t)16384 * 1024 * 4);  // time-major
  float*    XGtb  = (float*)   alloc((size_t)16384 * 1024 * 4);  // time-major
  _Float16* feats = (_Float16*)alloc((size_t)16384 * 512 * 2);
  _Float16* hid   = (_Float16*)alloc((size_t)16384 * 512 * 2);

  // weight conversion + transpose to f16 [N][K]
  cvt_transpose_kernel<<<(1024 * 256 + 255) / 256, 256, 0, stream>>>(wx_f, wxfT, 256, 1024);
  cvt_transpose_kernel<<<(1024 * 256 + 255) / 256, 256, 0, stream>>>(wx_b, wxbT, 256, 1024);
  cvt_transpose_kernel<<<(1024 * 256 + 255) / 256, 256, 0, stream>>>(wh_f, whfT, 256, 1024);
  cvt_transpose_kernel<<<(1024 * 256 + 255) / 256, 256, 0, stream>>>(wh_b, whbT, 256, 1024);
  cvt_transpose_kernel<<<(512 * 512 + 255) / 256, 256, 0, stream>>>(w1, w1T, 512, 512);

  gather_kernel<<<16384, 256, 0, stream>>>(input_ids, emb, X);

  gemm_xg_kernel<<<dim3(128, 16, 2), 256, 0, stream>>>(X, wxfT, b_f, wxbT, b_b, XGtf, XGtb);

  lstm_kernel<<<2, 1024, 0, stream>>>(XGtf, XGtb, whfT, whbT, feats);

  mlp_kernel<<<dim3(128, 8), 256, 0, stream>>>(feats, w1T, b1, hid);

  float* logits = (float*)d_out;
  logits_kernel<<<(16384 * 11 + 255) / 256, 256, 0, stream>>>(hid, w2, b2, logits);

  crf_kernel<<<1, 384, 0, stream>>>(logits, attn_mask, labels, trans,
                                    logits + 16384 * 11);
}
